// CLF_QP_Net_4483945857528
// MI455X (gfx1250) — matrix operand order repaired and checked by reading, not compiled, not run on hardware
//
#include <hip/hip_runtime.h>
#include <hip/hip_bf16.h>
#include <math.h>

// ---------------------------------------------------------------------------
// Problem dimensions (fixed by the reference)
// ---------------------------------------------------------------------------
#define B_ROWS 32768
#define N_IN   128
#define H_DIM  1024
#define C_DIM  32

typedef __attribute__((ext_vector_type(16))) __bf16 v16bf;
typedef __attribute__((ext_vector_type(8)))  __bf16 v8bf;
typedef __attribute__((ext_vector_type(8)))  float  v8f;
typedef int v4i_vs __attribute__((__vector_size__(16)));  // matches builtin proto

// ---------------------------------------------------------------------------
// Async global->LDS copy (CDNA5 GLOBAL_LOAD_ASYNC_TO_LDS_B128, ASYNCcnt).
// Builtin signature (from compiler diagnostic): pointers are int-vector(4)
// pointers in AS1 (global) / AS3 (LDS), plus imm offset and imm cpol.
// Falls back to a synchronous copy if the builtin is unavailable.
// ---------------------------------------------------------------------------
#if defined(__has_builtin)
#if __has_builtin(__builtin_amdgcn_global_load_async_to_lds_b128)
#define HAVE_ASYNC_LDS 1
#endif
#endif

#ifdef HAVE_ASYNC_LDS
#define ASYNC_COPY16(gsrc, ldst)                                       \
  __builtin_amdgcn_global_load_async_to_lds_b128(                      \
      (__attribute__((address_space(1))) v4i_vs*)(gsrc),               \
      (__attribute__((address_space(3))) v4i_vs*)(ldst), 0, 0)
__device__ __forceinline__ void waitAsync() {
#if __has_builtin(__builtin_amdgcn_s_wait_asynccnt)
  __builtin_amdgcn_s_wait_asynccnt(0);
#else
  asm volatile("s_wait_asynccnt 0x0" ::: "memory");
#endif
}
#else
#define ASYNC_COPY16(gsrc, ldst) (*(v8bf*)(ldst) = *(const v8bf*)(gsrc))
__device__ __forceinline__ void waitAsync() {}
#endif

// ---------------------------------------------------------------------------
// bf16 <-> f32 helpers (bit-exact, no reliance on __bf16 arithmetic support)
// ---------------------------------------------------------------------------
__device__ __forceinline__ __bf16 f2bf(float f) {
  union { float f; unsigned u; } v; v.f = f;
  unsigned r = v.u + 0x7FFFu + ((v.u >> 16) & 1u);   // round-to-nearest-even
  unsigned short h = (unsigned short)(r >> 16);
  __bf16 out; __builtin_memcpy(&out, &h, 2);
  return out;
}
__device__ __forceinline__ float bf2f(__bf16 b) {
  unsigned short h; __builtin_memcpy(&h, &b, 2);
  union { unsigned u; float f; } v; v.u = ((unsigned)h) << 16;
  return v.f;
}

// Branch-free tanh: 1 - 2/(exp(2x)+1).  Exact at +/-1 saturation, ~1ulp of
// fast-exp elsewhere (far above bf16 resolution).  v_exp_f32 + v_rcp_f32.
__device__ __forceinline__ float fast_tanh(float x) {
  float e = __expf(2.0f * x);                      // inf for large x -> ok
  return 1.0f - 2.0f * __builtin_amdgcn_rcpf(e + 1.0f);
}

// ---------------------------------------------------------------------------
// Elementwise prep kernels
// ---------------------------------------------------------------------------
__global__ void cvt_f32_to_bf16(const float* __restrict__ in,
                                __bf16* __restrict__ out, int n) {
  int i = blockIdx.x * blockDim.x + threadIdx.x;
  if (i < n) out[i] = f2bf(in[i]);
}

// out[c*rows + r] = bf16(in[r*cols + c])
__global__ void transpose_f32_to_bf16(const float* __restrict__ in,
                                      __bf16* __restrict__ out,
                                      int rows, int cols) {
  int i = blockIdx.x * blockDim.x + threadIdx.x;
  if (i < rows * cols) {
    int r = i / cols, c = i % cols;
    out[(size_t)c * rows + r] = f2bf(in[i]);
  }
}

__global__ void zero_f32(float* __restrict__ p, int n) {
  int i = blockIdx.x * blockDim.x + threadIdx.x;
  if (i < n) p[i] = 0.0f;
}

// ---------------------------------------------------------------------------
// NT GEMM via v_wmma_f32_16x16x32_bf16 with async-LDS double buffering:
//   D[M,Nn] = epilogue( A[M,K] * Bm[Nn,K]^T )
// Block = 256 threads = 8 wave32s arranged 2(M) x 4(N).
// Wave tile = 32x32 (2x2 WMMA tiles); block tile = 64 x 128; K-step 32.
// LDS tiles padded to 40-element row stride -> conflict-free ds_load_b128.
//
// Epilogues:
//   0: a1 = tanh(acc + bias[col])                 -> outBf
//   1: a2 = tanh(acc + bias[col]); s2=a2(1-a2^2)  -> outBf ; V += 0.5*a2^2
//      (ds_add_f32 column reduction, then one global atomic per row)
//   2: s1 = acc * (1 - a1^2), a1 read from aux    -> outBf (may alias aux)
//   3: plain f32 store                            -> outF32
// ---------------------------------------------------------------------------
#define BK   32
#define TSTR 40   // padded LDS row stride (elements)

template <int EPI>
__global__ __launch_bounds__(256)
void gemm_nt_wmma(const __bf16* __restrict__ A, const __bf16* __restrict__ Bm,
                  const float* __restrict__ bias,
                  __bf16* __restrict__ outBf, float* __restrict__ outF32,
                  const __bf16* __restrict__ aux, float* __restrict__ vAcc,
                  int K, int lda, int ldb, int ldo) {
  __shared__ __bf16 Asm[2][64 * TSTR];    // 2 x 5 KB
  __shared__ __bf16 Bsm[2][128 * TSTR];   // 2 x 10 KB
  __shared__ float vpart[64];

  const int t = threadIdx.x;
  const int lane = t & 31;
  const int wave = t >> 5;
  const int waveM = wave & 1;
  const int waveN = wave >> 1;
  const int l = lane & 15;     // row within A-frag / col within B-frag
  const int half = lane >> 4;  // lane half selects K sub-range
  const int blockRow = blockIdx.x * 64;
  const int colBlock = blockIdx.y * 128;

  // cooperative tile copy: 16B chunks, 4 chunks per 32-element K row
  const int crow = t >> 2;            // 0..63
  const int ccol = (t & 3) * 8;       // 0,8,16,24
  const __bf16* gA = A + (size_t)(blockRow + crow) * lda + ccol;
  const __bf16* gB0 = Bm + (size_t)(colBlock + crow) * ldb + ccol;
  const __bf16* gB1 = gB0 + (size_t)64 * ldb;

  v8f acc[2][2] = {};

  const int nk = K / BK;
  // prefetch tile 0
  ASYNC_COPY16(gA, &Asm[0][crow * TSTR + ccol]);
  ASYNC_COPY16(gB0, &Bsm[0][crow * TSTR + ccol]);
  ASYNC_COPY16(gB1, &Bsm[0][(crow + 64) * TSTR + ccol]);
  waitAsync();
  __syncthreads();

  for (int kt = 0; kt < nk; ++kt) {
    const int buf = kt & 1;
    if (kt + 1 < nk) {  // prefetch next K-tile into the other buffer
      const int ko = (kt + 1) * BK;
      ASYNC_COPY16(gA + ko, &Asm[buf ^ 1][crow * TSTR + ccol]);
      ASYNC_COPY16(gB0 + ko, &Bsm[buf ^ 1][crow * TSTR + ccol]);
      ASYNC_COPY16(gB1 + ko, &Bsm[buf ^ 1][(crow + 64) * TSTR + ccol]);
    }

    // fragments from LDS
    v16bf afrag[2], bfrag[2];
#pragma unroll
    for (int mi = 0; mi < 2; ++mi) {
      // A 16x32 layout: lane half 0 -> K {0..7,16..23}; half 1 -> {8..15,24..31}
      const __bf16* pa = &Asm[buf][(waveM * 32 + mi * 16 + l) * TSTR + half * 8];
      v8bf lo = *(const v8bf*)pa;
      v8bf hi = *(const v8bf*)(pa + 16);
      afrag[mi] = __builtin_shufflevector(lo, hi, 0, 1, 2, 3, 4, 5, 6, 7, 8, 9,
                                          10, 11, 12, 13, 14, 15);
    }
#pragma unroll
    for (int ni = 0; ni < 2; ++ni) {
      // B 32x16 layout: lane = column, half 0 -> K {0..7,16..23}; half 1 -> {8..15,24..31}
      const __bf16* pb = &Bsm[buf][(waveN * 32 + ni * 16 + l) * TSTR + half * 8];
      v8bf lo = *(const v8bf*)pb;
      v8bf hi = *(const v8bf*)(pb + 16);
      bfrag[ni] = __builtin_shufflevector(lo, hi, 0, 1, 2, 3, 4, 5, 6, 7, 8, 9,
                                          10, 11, 12, 13, 14, 15);
    }
#pragma unroll
    for (int mi = 0; mi < 2; ++mi)
#pragma unroll
      for (int ni = 0; ni < 2; ++ni)
        acc[mi][ni] = __builtin_amdgcn_wmma_f32_16x16x32_bf16(
            false, afrag[mi], false, bfrag[ni], (short)0, acc[mi][ni], false,
            false);

    waitAsync();      // next tile resident in LDS
    __syncthreads();  // all waves done reading `buf` before it is refilled
  }

  if (EPI == 1) {
    if (t < 64) vpart[t] = 0.0f;
    __syncthreads();
  }

  const int rowBase = blockRow + waveM * 32;
  const int colBase = colBlock + waveN * 32;
  // C/D layout: VGPR r -> M = r + 8*half, N = lane&15
#pragma unroll
  for (int mi = 0; mi < 2; ++mi) {
#pragma unroll
    for (int ni = 0; ni < 2; ++ni) {
#pragma unroll
      for (int r = 0; r < 8; ++r) {
        const int row = rowBase + mi * 16 + half * 8 + r;
        const int col = colBase + ni * 16 + l;
        const float v = acc[mi][ni][r];
        if (EPI == 0) {
          outBf[(size_t)row * ldo + col] = f2bf(fast_tanh(v + bias[col]));
        } else if (EPI == 1) {
          float a2 = fast_tanh(v + bias[col]);
          float s2 = a2 * (1.0f - a2 * a2);
          outBf[(size_t)row * ldo + col] = f2bf(s2);
          atomicAdd(&vpart[row - blockRow], a2 * a2);  // ds_add_f32
        } else if (EPI == 2) {
          float a1 = bf2f(aux[(size_t)row * ldo + col]);
          float s1 = v * (1.0f - a1 * a1);
          outBf[(size_t)row * ldo + col] = f2bf(s1);  // alias-safe: 1 thr/coord
        } else {
          outF32[(size_t)row * ldo + col] = v;
        }
      }
    }
  }

  if (EPI == 1) {
    __syncthreads();
    if (t < 64) atomicAdd(&vAcc[blockRow + t], 0.5f * vpart[t]);
  }
}

// ---------------------------------------------------------------------------
// Final fused kernel: one wave32 per batch row (f32 throughout).
//   L_f = gradV . (A_f x) ; L_g = Gmat^T gradV ; u_nom = -K x
//   s = L_f + V + L_g.u_nom ; r = relu(s)/(1+100*||L_g||^2)
//   u = u_nom - 100 r L_g ; Vdot = L_f + L_g.u
// Outputs concatenated: u[B*C] | r[B] | V[B] | Vdot[B]
// ---------------------------------------------------------------------------
__global__ __launch_bounds__(256)
void clf_qp_final(const float* __restrict__ x, const float* __restrict__ gradV,
                  const float* __restrict__ A_f, const float* __restrict__ Gm,
                  const float* __restrict__ Km, const float* __restrict__ vAcc,
                  float* __restrict__ out) {
  __shared__ float xs[8][N_IN];
  __shared__ float gs[8][N_IN];
  const int wave = threadIdx.x >> 5;
  const int lane = threadIdx.x & 31;
  const int row = blockIdx.x * 8 + wave;

  ((float4*)xs[wave])[lane] = ((const float4*)(x + (size_t)row * N_IN))[lane];
  ((float4*)gs[wave])[lane] =
      ((const float4*)(gradV + (size_t)row * N_IN))[lane];
  __syncthreads();

  float lf = 0.0f;
  for (int i = lane; i < N_IN; i += 32) {
    const float* ar = A_f + (size_t)i * N_IN;
    float y = 0.0f;
#pragma unroll 4
    for (int n = 0; n < N_IN; ++n) y = fmaf(ar[n], xs[wave][n], y);
    lf = fmaf(gs[wave][i], y, lf);
  }
  for (int off = 16; off > 0; off >>= 1) lf += __shfl_down(lf, off, 32);
  lf = __shfl(lf, 0, 32);

  const int c = lane;  // C == 32 == wave width
  float lg = 0.0f, un = 0.0f;
  const float* kr = Km + (size_t)c * N_IN;
  for (int n = 0; n < N_IN; ++n) {
    lg = fmaf(gs[wave][n], Gm[(size_t)n * C_DIM + c], lg);
    un = fmaf(-kr[n], xs[wave][n], un);
  }

  const float V = vAcc[row];
  float s_red = lg * un;
  float q_red = lg * lg;
  for (int off = 16; off > 0; off >>= 1) {
    s_red += __shfl_down(s_red, off, 32);
    q_red += __shfl_down(q_red, off, 32);
  }
  s_red = __shfl(s_red, 0, 32);
  q_red = __shfl(q_red, 0, 32);

  const float s = lf + 1.0f * V + s_red;  // CLF_LAMBDA = 1
  const float r = fmaxf(s, 0.0f) / (1.0f + 100.0f * q_red);
  const float u = un - 100.0f * r * lg;

  float vd = lg * u;
  for (int off = 16; off > 0; off >>= 1) vd += __shfl_down(vd, off, 32);

  out[(size_t)row * C_DIM + c] = u;
  if (lane == 0) {
    const size_t BB = B_ROWS;
    out[BB * C_DIM + row] = r;
    out[BB * C_DIM + BB + row] = V;
    out[BB * C_DIM + 2 * BB + row] = lf + vd;
  }
}

// ---------------------------------------------------------------------------
extern "C" void kernel_launch(void* const* d_in, const int* in_sizes, int n_in,
                              void* d_out, int out_size, void* d_ws,
                              size_t ws_size, hipStream_t stream) {
  (void)in_sizes; (void)n_in; (void)out_size; (void)ws_size;
  const float* x  = (const float*)d_in[0];
  const float* W1 = (const float*)d_in[1];
  const float* b1 = (const float*)d_in[2];
  const float* W2 = (const float*)d_in[3];
  const float* b2 = (const float*)d_in[4];
  const float* Af = (const float*)d_in[5];
  const float* Gm = (const float*)d_in[6];
  const float* Km = (const float*)d_in[7];
  float* out = (float*)d_out;

  char* ws = (char*)d_ws;
  auto carve = [&](size_t bytes) {
    char* p = ws;
    ws += (bytes + 255) & ~(size_t)255;
    return p;
  };
  __bf16* x_bf   = (__bf16*)carve((size_t)B_ROWS * N_IN * 2);
  __bf16* W1_bf  = (__bf16*)carve((size_t)H_DIM * N_IN * 2);
  __bf16* W2_bf  = (__bf16*)carve((size_t)H_DIM * H_DIM * 2);
  __bf16* W2T_bf = (__bf16*)carve((size_t)H_DIM * H_DIM * 2);
  __bf16* W1T_bf = (__bf16*)carve((size_t)N_IN * H_DIM * 2);
  __bf16* a1_bf  = (__bf16*)carve((size_t)B_ROWS * H_DIM * 2);  // reused as s1
  __bf16* s2_bf  = (__bf16*)carve((size_t)B_ROWS * H_DIM * 2);
  float*  gradV  = (float*) carve((size_t)B_ROWS * N_IN * 4);
  float*  vAcc   = (float*) carve((size_t)B_ROWS * 4);

  const int T = 256;
  auto nb = [&](size_t n) { return (unsigned)((n + T - 1) / T); };

  cvt_f32_to_bf16<<<nb((size_t)B_ROWS * N_IN), T, 0, stream>>>(x, x_bf,
                                                               B_ROWS * N_IN);
  cvt_f32_to_bf16<<<nb((size_t)H_DIM * N_IN), T, 0, stream>>>(W1, W1_bf,
                                                              H_DIM * N_IN);
  cvt_f32_to_bf16<<<nb((size_t)H_DIM * H_DIM), T, 0, stream>>>(W2, W2_bf,
                                                               H_DIM * H_DIM);
  transpose_f32_to_bf16<<<nb((size_t)H_DIM * H_DIM), T, 0, stream>>>(
      W2, W2T_bf, H_DIM, H_DIM);
  transpose_f32_to_bf16<<<nb((size_t)H_DIM * N_IN), T, 0, stream>>>(
      W1, W1T_bf, H_DIM, N_IN);
  zero_f32<<<nb(B_ROWS), T, 0, stream>>>(vAcc, B_ROWS);

  const dim3 gBig(B_ROWS / 64, H_DIM / 128);  // 512 x 8
  const dim3 gN(B_ROWS / 64, N_IN / 128);     // 512 x 1

  // a1 = tanh(x W1^T + b1)
  gemm_nt_wmma<0><<<gBig, T, 0, stream>>>(x_bf, W1_bf, b1, a1_bf, nullptr,
                                          nullptr, nullptr, N_IN, N_IN, N_IN,
                                          H_DIM);
  // a2 = tanh(a1 W2^T + b2); s2 = a2(1-a2^2); V += 0.5 a2^2
  gemm_nt_wmma<1><<<gBig, T, 0, stream>>>(a1_bf, W2_bf, b2, s2_bf, nullptr,
                                          nullptr, vAcc, H_DIM, H_DIM, H_DIM,
                                          H_DIM);
  // t = s2 W2 (NT against W2^T); s1 = t(1-a1^2) written over a1
  gemm_nt_wmma<2><<<gBig, T, 0, stream>>>(s2_bf, W2T_bf, nullptr, a1_bf,
                                          nullptr, a1_bf, nullptr, H_DIM,
                                          H_DIM, H_DIM, H_DIM);
  // grad_V = s1 W1 (NT against W1^T), f32 out
  gemm_nt_wmma<3><<<gN, T, 0, stream>>>(a1_bf, W1T_bf, nullptr, nullptr, gradV,
                                        nullptr, nullptr, H_DIM, H_DIM, H_DIM,
                                        N_IN);
  // fused dynamics + CLF-QP tail
  clf_qp_final<<<B_ROWS / 8, T, 0, stream>>>(x, gradV, Af, Gm, Km, vAcc, out);
}
